// RnnSenderGS_23192823399180
// MI455X (gfx1250) — matrix operand order repaired and checked by reading, not compiled, not run on hardware
//
#include <hip/hip_runtime.h>
#include <hip/hip_bf16.h>
#include <math.h>

// ---------------------------------------------------------------------------
// Problem constants (match reference)
// ---------------------------------------------------------------------------
#define B_SZ   128
#define DIN    512
#define E_SZ   256
#define H_SZ   512
#define V_SZ   32000
#define L_SZ   16
#define LOG2E_F 1.4426950408889634f

typedef __bf16 bf16;
typedef __bf16 v16bf __attribute__((ext_vector_type(16)));
typedef __bf16 v8bf  __attribute__((ext_vector_type(8)));
typedef float  v8f   __attribute__((ext_vector_type(8)));

#define WMMA_BF16(a, b, c) \
    __builtin_amdgcn_wmma_f32_16x16x32_bf16(false, (a), false, (b), (short)0, (c), false, false)

__device__ __forceinline__ bf16 f2bf(float f) { return (bf16)f; }

// A-operand (16x32, bf16, row-major source with leading dim ld):
// lane<16: M=lane, K in {0..7} U {16..23}; lane>=16: M=lane-16, K in {8..15} U {24..31}
__device__ __forceinline__ v16bf loadA16x32(const bf16* A, int ld) {
    int lane = threadIdx.x & 31;
    int m  = lane & 15;
    int kb = (lane >> 4) << 3;            // 0 or 8
    const bf16* p = A + m * ld + kb;
    v8bf lo = *(const v8bf*)(p);          // K = kb .. kb+7
    v8bf hi = *(const v8bf*)(p + 16);     // K = kb+16 .. kb+23
    v16bf r;
#pragma unroll
    for (int i = 0; i < 8; ++i) { r[i] = lo[i]; r[8 + i] = hi[i]; }
    return r;
}

// B-operand (32x16, bf16). We compute X @ W^T, so B[k][n] = W[n][k] with W
// row-major (leading dim ld). lane<16: N=lane, K in {0..7} U {16..23}; lane>=16: N=lane-16, K in {8..15} U {24..31}
__device__ __forceinline__ v16bf loadB32x16(const bf16* W, int ld) {
    int lane = threadIdx.x & 31;
    int n  = lane & 15;
    int kb = (lane >> 4) << 3;            // 0 or 8
    const bf16* p = W + n * ld + kb;
    v8bf lo = *(const v8bf*)(p);
    v8bf hi = *(const v8bf*)(p + 16);
    v16bf r;
#pragma unroll
    for (int i = 0; i < 8; ++i) { r[i] = lo[i]; r[8 + i] = hi[i]; }
    return r;
}

// C/D layout: VGPR v: lanes 0-15 -> (M=v,   N=lane),
//                     lanes16-31 -> (M=v+8, N=lane-16)

// ---------------------------------------------------------------------------
// f32 -> bf16 bulk conversion (weights done once; fits in L2 afterwards)
// ---------------------------------------------------------------------------
__global__ void cvt_bf16_kernel(const float* __restrict__ src, bf16* __restrict__ dst, long n) {
    long i = (long)blockIdx.x * blockDim.x + threadIdx.x;
    long stride = (long)gridDim.x * blockDim.x;
    for (; i < n; i += stride) dst[i] = f2bf(src[i]);
}

// ---------------------------------------------------------------------------
// h0 = x @ W_agent^T + b_agent      (one wave per 16x16 tile, K=512)
// ---------------------------------------------------------------------------
__global__ __launch_bounds__(32) void init_h_kernel(const bf16* __restrict__ xB,
                                                    const bf16* __restrict__ WagB,
                                                    const float* __restrict__ bag,
                                                    float* __restrict__ hF,
                                                    bf16*  __restrict__ hBout) {
    int m0 = blockIdx.x * 16, n0 = blockIdx.y * 16;
    v8f acc = {};
    for (int k = 0; k < DIN; k += 32) {
        v16bf a = loadA16x32(xB + m0 * DIN + k, DIN);
        v16bf b = loadB32x16(WagB + n0 * DIN + k, DIN);
        acc = WMMA_BF16(a, b, acc);
    }
    int lane = threadIdx.x & 31;
    int n = lane & 15, mb = (lane >> 4) * 8;
    int col = n0 + n;
    float bias = bag[col];
#pragma unroll
    for (int v = 0; v < 8; ++v) {
        int row = m0 + mb + v;
        float val = acc[v] + bias;
        hF[row * H_SZ + col]    = val;
        hBout[row * H_SZ + col] = f2bf(val);
    }
}

// e0 = broadcast(sos)
__global__ void init_e_kernel(const float* __restrict__ sos, bf16* __restrict__ eB) {
    int b = blockIdx.x, j = threadIdx.x;       // grid(128), block(256)
    eB[b * E_SZ + j] = f2bf(sos[j]);
}

// ---------------------------------------------------------------------------
// GRU step: gx = e@Wih^T, gh = h@Whh^T, gates, h_new.
// One wave per 16x16 tile of H; computes 6 WMMA accumulators (3 gates x 2).
// h bf16 is double-buffered (gh GEMM reads ALL columns of hBin).
// ---------------------------------------------------------------------------
__global__ __launch_bounds__(32) void gru_kernel(const bf16* __restrict__ eB,
                                                 const bf16* __restrict__ hBin,
                                                 const bf16* __restrict__ WihB,
                                                 const bf16* __restrict__ WhhB,
                                                 const float* __restrict__ bih,
                                                 const float* __restrict__ bhh,
                                                 float* __restrict__ hF,
                                                 bf16*  __restrict__ hBout) {
    int m0 = blockIdx.x * 16, j0 = blockIdx.y * 16;
    v8f gx0 = {}, gx1 = {}, gx2 = {}, gh0 = {}, gh1 = {}, gh2 = {};
    for (int k = 0; k < E_SZ; k += 32) {
        v16bf a = loadA16x32(eB + m0 * E_SZ + k, E_SZ);
        gx0 = WMMA_BF16(a, loadB32x16(WihB + (j0            ) * E_SZ + k, E_SZ), gx0);
        gx1 = WMMA_BF16(a, loadB32x16(WihB + (j0 +     H_SZ ) * E_SZ + k, E_SZ), gx1);
        gx2 = WMMA_BF16(a, loadB32x16(WihB + (j0 + 2 * H_SZ ) * E_SZ + k, E_SZ), gx2);
    }
    for (int k = 0; k < H_SZ; k += 32) {
        v16bf a = loadA16x32(hBin + m0 * H_SZ + k, H_SZ);
        gh0 = WMMA_BF16(a, loadB32x16(WhhB + (j0            ) * H_SZ + k, H_SZ), gh0);
        gh1 = WMMA_BF16(a, loadB32x16(WhhB + (j0 +     H_SZ ) * H_SZ + k, H_SZ), gh1);
        gh2 = WMMA_BF16(a, loadB32x16(WhhB + (j0 + 2 * H_SZ ) * H_SZ + k, H_SZ), gh2);
    }
    int lane = threadIdx.x & 31;
    int n = lane & 15, mb = (lane >> 4) * 8;
    int j = j0 + n;
    float bi0 = bih[j], bi1 = bih[j + H_SZ], bi2 = bih[j + 2 * H_SZ];
    float bh0 = bhh[j], bh1 = bhh[j + H_SZ], bh2 = bhh[j + 2 * H_SZ];
#pragma unroll
    for (int v = 0; v < 8; ++v) {
        int row = m0 + mb + v;
        float r  = 1.0f / (1.0f + expf(-(gx0[v] + bi0 + gh0[v] + bh0)));
        float z  = 1.0f / (1.0f + expf(-(gx1[v] + bi1 + gh1[v] + bh1)));
        float nn = tanhf(gx2[v] + bi2 + r * (gh2[v] + bh2));
        float hold = hF[row * H_SZ + j];
        float hnew = (1.0f - z) * nn + z * hold;
        hF[row * H_SZ + j]    = hnew;
        hBout[row * H_SZ + j] = f2bf(hnew);
    }
}

// ---------------------------------------------------------------------------
// Raw logits: h_new @ Wout^T + bout  ->  d_out logits slot [B, L, V].
// Block = 4 waves; each wave owns a 16x64 strip (4 accumulators sharing A).
// ---------------------------------------------------------------------------
__global__ __launch_bounds__(128) void logits_kernel(const bf16* __restrict__ hB,
                                                     const bf16* __restrict__ WoutB,
                                                     const float* __restrict__ bout,
                                                     float* __restrict__ logits_out,
                                                     int l) {
    int m0 = blockIdx.x * 16;
    int wave = threadIdx.x >> 5;
    int n0 = blockIdx.y * 256 + wave * 64;
    v8f a0 = {}, a1 = {}, a2 = {}, a3 = {};
    for (int k = 0; k < H_SZ; k += 32) {
        v16bf a  = loadA16x32(hB + m0 * H_SZ + k, H_SZ);
        v16bf b0 = loadB32x16(WoutB + (long)(n0 +  0) * H_SZ + k, H_SZ);
        v16bf b1 = loadB32x16(WoutB + (long)(n0 + 16) * H_SZ + k, H_SZ);
        v16bf b2 = loadB32x16(WoutB + (long)(n0 + 32) * H_SZ + k, H_SZ);
        v16bf b3 = loadB32x16(WoutB + (long)(n0 + 48) * H_SZ + k, H_SZ);
        a0 = WMMA_BF16(a, b0, a0);
        a1 = WMMA_BF16(a, b1, a1);
        a2 = WMMA_BF16(a, b2, a2);
        a3 = WMMA_BF16(a, b3, a3);
    }
    int lane = threadIdx.x & 31;
    int n = lane & 15, mb = (lane >> 4) * 8;
    v8f accs[4] = {a0, a1, a2, a3};
#pragma unroll
    for (int t = 0; t < 4; ++t) {
        int col = n0 + 16 * t + n;
        float bias = bout[col];
#pragma unroll
        for (int v = 0; v < 8; ++v) {
            int row = m0 + mb + v;
            logits_out[((long)row * L_SZ + l) * V_SZ + col] = accs[t][v] + bias;
        }
    }
}

// ---------------------------------------------------------------------------
// Per-row softmax statistics (two passes over V=32000) + entropy.
// stats[b] = { logZ(logits), logZ(logits + gumbel) }
// ent = LOG2E * (ln s1 - t1/s1),  s1 = sum e^(x-m1), t1 = sum (x-m1) e^(x-m1)
// ---------------------------------------------------------------------------
__global__ __launch_bounds__(256) void softmax_reduce_kernel(const float* __restrict__ logits_raw,
                                                             const float* __restrict__ gumbels,
                                                             float* __restrict__ stats,
                                                             float* __restrict__ ent_out,
                                                             int l) {
    int b = blockIdx.x, tid = threadIdx.x;
    const float* row = logits_raw + ((long)b * L_SZ + l) * V_SZ;
    const float* g   = gumbels    + ((long)l * B_SZ + b) * V_SZ;
    __shared__ float sm[256];

    float m1 = -INFINITY, m2 = -INFINITY;
    for (int v = tid; v < V_SZ; v += 256) {
        float x = row[v];
        m1 = fmaxf(m1, x);
        m2 = fmaxf(m2, x + g[v]);
    }
    sm[tid] = m1; __syncthreads();
    for (int s = 128; s > 0; s >>= 1) { if (tid < s) sm[tid] = fmaxf(sm[tid], sm[tid + s]); __syncthreads(); }
    m1 = sm[0]; __syncthreads();
    sm[tid] = m2; __syncthreads();
    for (int s = 128; s > 0; s >>= 1) { if (tid < s) sm[tid] = fmaxf(sm[tid], sm[tid + s]); __syncthreads(); }
    m2 = sm[0]; __syncthreads();

    float s1 = 0.f, t1 = 0.f, s2 = 0.f;
    for (int v = tid; v < V_SZ; v += 256) {
        float x = row[v];
        float d = x - m1;
        float e1 = expf(d);
        s1 += e1;
        t1 += d * e1;
        s2 += expf(x + g[v] - m2);
    }
    sm[tid] = s1; __syncthreads();
    for (int s = 128; s > 0; s >>= 1) { if (tid < s) sm[tid] += sm[tid + s]; __syncthreads(); }
    s1 = sm[0]; __syncthreads();
    sm[tid] = t1; __syncthreads();
    for (int s = 128; s > 0; s >>= 1) { if (tid < s) sm[tid] += sm[tid + s]; __syncthreads(); }
    t1 = sm[0]; __syncthreads();
    sm[tid] = s2; __syncthreads();
    for (int s = 128; s > 0; s >>= 1) { if (tid < s) sm[tid] += sm[tid + s]; __syncthreads(); }
    s2 = sm[0];

    if (tid == 0) {
        float ls1 = logf(s1);
        stats[b * 2 + 0] = m1 + ls1;
        stats[b * 2 + 1] = m2 + logf(s2);
        ent_out[b * (L_SZ + 1) + l] = LOG2E_F * (ls1 - t1 / s1);
    }
}

// ---------------------------------------------------------------------------
// Elementwise: norm_logits (in place), sample -> seq slot (+ bf16 copy for
// the embedding GEMM).
// ---------------------------------------------------------------------------
__global__ __launch_bounds__(256) void softmax_map_kernel(float* __restrict__ logits_io,
                                                          const float* __restrict__ gumbels,
                                                          const float* __restrict__ stats,
                                                          float* __restrict__ seq_out,
                                                          bf16*  __restrict__ sampleB,
                                                          int l) {
    int b = blockIdx.x;
    int v = blockIdx.y * 256 + threadIdx.x;   // grid(128, 125)
    long li = ((long)b * L_SZ + l) * V_SZ + v;
    float x = logits_io[li];
    float g = gumbels[((long)l * B_SZ + b) * V_SZ + v];
    float lz1 = stats[b * 2 + 0];
    float lz2 = stats[b * 2 + 1];
    logits_io[li] = x - lz1;                                  // log_softmax
    float smp = expf(x + g - lz2);                            // gumbel-softmax sample
    seq_out[((long)b * (L_SZ + 1) + l) * V_SZ + v] = smp;
    sampleB[(long)b * V_SZ + v] = f2bf(smp);
}

// ---------------------------------------------------------------------------
// e_new = sample @ Wemb^T + bemb   (K = 32000 -> 1000 WMMA per wave)
// ---------------------------------------------------------------------------
__global__ __launch_bounds__(128) void embed_kernel(const bf16* __restrict__ sampleB,
                                                    const bf16* __restrict__ WembB,
                                                    const float* __restrict__ bemb,
                                                    bf16* __restrict__ eB) {
    int m0 = blockIdx.x * 16;
    int wave = threadIdx.x >> 5;
    int n0 = blockIdx.y * 64 + wave * 16;     // grid(8,4) x 4 waves -> E=256
    v8f acc = {};
    for (int k = 0; k < V_SZ; k += 32) {
        v16bf a = loadA16x32(sampleB + (long)m0 * V_SZ + k, V_SZ);
        v16bf b = loadB32x16(WembB + (long)n0 * V_SZ + k, V_SZ);
        acc = WMMA_BF16(a, b, acc);
    }
    int lane = threadIdx.x & 31;
    int n = lane & 15, mb = (lane >> 4) * 8;
    int col = n0 + n;
    float bias = bemb[col];
#pragma unroll
    for (int v = 0; v < 8; ++v) {
        int row = m0 + mb + v;
        eB[row * E_SZ + col] = f2bf(acc[v] + bias);
    }
}

// ---------------------------------------------------------------------------
// EOS frame + trailing zero entropy column
// ---------------------------------------------------------------------------
__global__ __launch_bounds__(256) void tail_kernel(float* __restrict__ seq_out,
                                                   float* __restrict__ ent_out) {
    int b = blockIdx.x;
    int v = blockIdx.y * 256 + threadIdx.x;   // grid(128, 125)
    seq_out[((long)b * (L_SZ + 1) + L_SZ) * V_SZ + v] = (v == 0) ? 1.0f : 0.0f;
    if (blockIdx.y == 0 && threadIdx.x == 0)
        ent_out[b * (L_SZ + 1) + L_SZ] = 0.0f;
}

// ---------------------------------------------------------------------------
// Host launcher
// ---------------------------------------------------------------------------
extern "C" void kernel_launch(void* const* d_in, const int* in_sizes, int n_in,
                              void* d_out, int out_size, void* d_ws, size_t ws_size,
                              hipStream_t stream) {
    (void)in_sizes; (void)n_in; (void)out_size; (void)ws_size;
    const float* x       = (const float*)d_in[0];
    const float* gumbels = (const float*)d_in[1];
    const float* W_agent = (const float*)d_in[2];
    const float* b_agent = (const float*)d_in[3];
    const float* Wih     = (const float*)d_in[4];
    const float* bih     = (const float*)d_in[5];
    const float* Whh     = (const float*)d_in[6];
    const float* bhh     = (const float*)d_in[7];
    const float* Wout    = (const float*)d_in[8];
    const float* bout    = (const float*)d_in[9];
    const float* Wemb    = (const float*)d_in[10];
    const float* bemb    = (const float*)d_in[11];
    const float* sos     = (const float*)d_in[12];

    float* out        = (float*)d_out;
    float* seq_out    = out;                                          // [128,17,32000]
    float* logits_out = out + (size_t)B_SZ * (L_SZ + 1) * V_SZ;       // [128,16,32000]
    float* ent_out    = logits_out + (size_t)B_SZ * L_SZ * V_SZ;      // [128,17]

    // workspace carve-out (256B aligned slots); total ~61 MB
    char* ws = (char*)d_ws;
    size_t off = 0;
    auto carve = [&](size_t bytes) -> void* {
        void* p = ws + off;
        off += (bytes + 255) & ~(size_t)255;
        return p;
    };
    bf16* WoutB   = (bf16*)carve((size_t)V_SZ * H_SZ * 2);
    bf16* WembB   = (bf16*)carve((size_t)E_SZ * V_SZ * 2);
    bf16* WihB    = (bf16*)carve((size_t)3 * H_SZ * E_SZ * 2);
    bf16* WhhB    = (bf16*)carve((size_t)3 * H_SZ * H_SZ * 2);
    bf16* WagB    = (bf16*)carve((size_t)H_SZ * DIN * 2);
    bf16* xB      = (bf16*)carve((size_t)B_SZ * DIN * 2);
    bf16* sampleB = (bf16*)carve((size_t)B_SZ * V_SZ * 2);
    float* hF     = (float*)carve((size_t)B_SZ * H_SZ * 4);
    bf16* hB0     = (bf16*)carve((size_t)B_SZ * H_SZ * 2);
    bf16* hB1     = (bf16*)carve((size_t)B_SZ * H_SZ * 2);
    bf16* eB      = (bf16*)carve((size_t)B_SZ * E_SZ * 2);
    float* stats  = (float*)carve((size_t)B_SZ * 2 * 4);

    auto cvt = [&](const float* s, bf16* d, long n) {
        int blocks = (int)((n + 255) / 256);
        if (blocks > 16384) blocks = 16384;
        cvt_bf16_kernel<<<blocks, 256, 0, stream>>>(s, d, n);
    };
    cvt(Wout,    WoutB, (long)V_SZ * H_SZ);
    cvt(Wemb,    WembB, (long)E_SZ * V_SZ);
    cvt(Wih,     WihB,  (long)3 * H_SZ * E_SZ);
    cvt(Whh,     WhhB,  (long)3 * H_SZ * H_SZ);
    cvt(W_agent, WagB,  (long)H_SZ * DIN);
    cvt(x,       xB,    (long)B_SZ * DIN);

    init_h_kernel<<<dim3(B_SZ / 16, H_SZ / 16), 32, 0, stream>>>(xB, WagB, b_agent, hF, hB0);
    init_e_kernel<<<B_SZ, E_SZ, 0, stream>>>(sos, eB);

    const bf16* hRead = hB0;
    bf16* hWrite = hB1;
    for (int l = 0; l < L_SZ; ++l) {
        gru_kernel<<<dim3(B_SZ / 16, H_SZ / 16), 32, 0, stream>>>(
            eB, hRead, WihB, WhhB, bih, bhh, hF, hWrite);
        logits_kernel<<<dim3(B_SZ / 16, V_SZ / 256), 128, 0, stream>>>(
            hWrite, WoutB, bout, logits_out, l);
        softmax_reduce_kernel<<<B_SZ, 256, 0, stream>>>(
            logits_out, gumbels, stats, ent_out, l);
        softmax_map_kernel<<<dim3(B_SZ, V_SZ / 256), 256, 0, stream>>>(
            logits_out, gumbels, stats, seq_out, sampleB, l);
        embed_kernel<<<dim3(B_SZ / 16, E_SZ / 64), 128, 0, stream>>>(
            sampleB, WembB, bemb, eB);
        // swap h double-buffer
        const bf16* tmp = hRead;
        hRead = hWrite;
        hWrite = (bf16*)tmp;
    }

    tail_kernel<<<dim3(B_SZ, V_SZ / 256), 256, 0, stream>>>(seq_out, ent_out);
}
